// MambaBlock_9431748182252
// MI455X (gfx1250) — compile-verified, matrix operand order checked by reading, not run on hardware
//
#include <hip/hip_runtime.h>
#include <hip/hip_bf16.h>

#define L_SEQ   4096
#define DM      1024
#define DI      2048
#define DTR     64
#define DS      16
#define KCONV   4
#define XDBL_N  (DTR + 2*DS)   // 96

typedef __attribute__((ext_vector_type(16))) __bf16 v16bf;
typedef __attribute__((ext_vector_type(8)))  float  v8f;
typedef __attribute__((ext_vector_type(4)))  int    v4i;
typedef unsigned short ushort_t;

typedef __attribute__((address_space(1))) v4i gv4i;  // global int4
typedef __attribute__((address_space(3))) v4i lv4i;  // LDS int4

#if defined(__has_builtin)
#  if __has_builtin(__builtin_amdgcn_global_load_async_to_lds_b128)
#    define HAVE_ASYNC_LDS 1
#  endif
#endif
#ifndef HAVE_ASYNC_LDS
#  define HAVE_ASYNC_LDS 0
#endif

union FragBF {
    uint4 u[2];   // 32 bytes = 16 bf16
    v16bf v;
};

__device__ __forceinline__ float bf2f(ushort_t h) {
    unsigned int u = ((unsigned int)h) << 16;
    return __uint_as_float(u);
}
__device__ __forceinline__ ushort_t f2bf(float f) {
    unsigned int u = __float_as_uint(f);
    unsigned int r = ((u >> 16) & 1u) + 0x7FFFu;   // round to nearest even
    return (ushort_t)((u + r) >> 16);
}

// 16-byte global -> LDS copy: async DMA path on CDNA5, sync fallback otherwise.
__device__ __forceinline__ void copy16_g2l(const ushort_t* g, ushort_t* l) {
#if HAVE_ASYNC_LDS
    __builtin_amdgcn_global_load_async_to_lds_b128((gv4i*)g, (lv4i*)l, 0, 0);
#else
    *(uint4*)l = *(const uint4*)g;
#endif
}
__device__ __forceinline__ void wait_async_all() {
#if HAVE_ASYNC_LDS
#  if __has_builtin(__builtin_amdgcn_s_wait_asynccnt)
    __builtin_amdgcn_s_wait_asynccnt(0);
#  else
    asm volatile("s_wait_asynccnt 0x0" ::: "memory");
#  endif
#endif
}

// ---------------------------------------------------------------- f32 -> bf16
__global__ void k_f2bf(const float* __restrict__ in, ushort_t* __restrict__ out, int n) {
    for (int i = blockIdx.x * blockDim.x + threadIdx.x; i < n; i += gridDim.x * blockDim.x)
        out[i] = f2bf(in[i]);
}

// strided f32 -> bf16 (extract delta_r columns from x_dbl)
__global__ void k_f2bf_strided(const float* __restrict__ in, ushort_t* __restrict__ out,
                               int rows, int cols, int in_ld, int out_ld) {
    int n = rows * cols;
    for (int i = blockIdx.x * blockDim.x + threadIdx.x; i < n; i += gridDim.x * blockDim.x) {
        int r = i / cols, c = i - r * cols;
        out[(size_t)r * out_ld + c] = f2bf(in[(size_t)r * in_ld + c]);
    }
}

// ---------------------------------------------------------------- LayerNorm -> bf16
__global__ void k_layernorm(const float* __restrict__ x, const float* __restrict__ w,
                            const float* __restrict__ b, ushort_t* __restrict__ out) {
    const int row = blockIdx.x;
    const float* xr = x + (size_t)row * DM;
    float s = 0.f, s2 = 0.f;
    for (int i = threadIdx.x; i < DM; i += 256) {
        float v = xr[i];
        s += v; s2 += v * v;
    }
    #pragma unroll
    for (int o = 16; o >= 1; o >>= 1) {
        s  += __shfl_xor(s,  o, 32);
        s2 += __shfl_xor(s2, o, 32);
    }
    __shared__ float rs[8], rs2[8];
    const int wid = threadIdx.x >> 5, lane = threadIdx.x & 31;
    if (lane == 0) { rs[wid] = s; rs2[wid] = s2; }
    __syncthreads();
    float S = 0.f, S2 = 0.f;
    #pragma unroll
    for (int i = 0; i < 8; i++) { S += rs[i]; S2 += rs2[i]; }
    const float mu = S * (1.0f / DM);
    const float var = S2 * (1.0f / DM) - mu * mu;
    const float rstd = rsqrtf(var + 1e-5f);
    for (int i = threadIdx.x; i < DM; i += 256)
        out[(size_t)row * DM + i] = f2bf((xr[i] - mu) * rstd * w[i] + b[i]);
}

// ---------------------------------------------------------------- GEMM epilogue helper
// C/D layout: VGPR e -> row e + 8*(lane>=16), col = lane&15
template <int MODE>
__device__ __forceinline__ void gemm_epilogue(v8f acc[2][4], void* Cp, int ldc,
                                              int M, int N, int bm, int bn,
                                              int wm, int wn, int lane,
                                              const float* aux) {
    const int rsel = (lane >> 4) * 8;
    const int lcol = lane & 15;
    #pragma unroll
    for (int i = 0; i < 2; i++) {
        #pragma unroll
        for (int j = 0; j < 4; j++) {
            const int colg = bn + wn * 64 + j * 16 + lcol;
            #pragma unroll
            for (int e = 0; e < 8; e++) {
                const int rowg = bm + wm * 32 + i * 16 + rsel + e;
                if (rowg < M && colg < N) {
                    const float v = acc[i][j][e];
                    if (MODE == 0) {
                        ((float*)Cp)[(size_t)rowg * ldc + colg] = v;
                    } else if (MODE == 1) {
                        ((ushort_t*)Cp)[(size_t)rowg * ldc + colg] = f2bf(v);
                    } else if (MODE == 2) {
                        float t = v + aux[colg];
                        t = (t > 20.f) ? t : log1pf(__expf(t));
                        ((float*)Cp)[(size_t)rowg * ldc + colg] = t;
                    } else {
                        ((float*)Cp)[(size_t)rowg * ldc + colg] =
                            v + aux[(size_t)rowg * ldc + colg];
                    }
                }
            }
        }
    }
}

// ---------------------------------------------------------------- bf16 WMMA GEMM (full tiles)
// C[M,N] = A[M,K] * B[N,K]^T, M%128==0, N%128==0, K%32==0.
// Double-buffered LDS; global->LDS via ASYNCcnt-tracked async loads (CDNA5),
// overlapping next tile's DMA with current tile's 8 v_wmma_f32_16x16x32_bf16.
template <int MODE>
__global__ void k_gemm_bf16_dbuf(const ushort_t* __restrict__ A, int lda,
                                 const ushort_t* __restrict__ B, int ldb,
                                 void* __restrict__ Cp, int ldc,
                                 int M, int N, int K,
                                 const float* __restrict__ aux) {
    __shared__ ushort_t As[2][128 * 40];   // 40-half stride: conflict-free b128 reads
    __shared__ ushort_t Bs[2][128 * 40];

    const int tid  = threadIdx.x;
    const int lane = tid & 31;
    const int wid  = tid >> 5;
    const int wm   = wid & 3;        // 4 wave rows  -> 32 M each
    const int wn   = wid >> 2;       // 2 wave cols  -> 64 N each
    const int bm   = blockIdx.y * 128;
    const int bn   = blockIdx.x * 128;

    // staging: each thread moves 4x16B (A rows r0,r0+64 and B rows r0,r0+64 at col c8)
    const int r0 = tid >> 2;
    const int c8 = (tid & 3) << 3;
    const ushort_t* gA0 = A + (size_t)(bm + r0) * lda + c8;
    const ushort_t* gA1 = gA0 + (size_t)64 * lda;
    const ushort_t* gB0 = B + (size_t)(bn + r0) * ldb + c8;
    const ushort_t* gB1 = gB0 + (size_t)64 * ldb;
    const int l0 = r0 * 40 + c8;
    const int l1 = (r0 + 64) * 40 + c8;

    auto stage = [&](int buf, int k0) {
        copy16_g2l(gA0 + k0, &As[buf][l0]);
        copy16_g2l(gA1 + k0, &As[buf][l1]);
        copy16_g2l(gB0 + k0, &Bs[buf][l0]);
        copy16_g2l(gB1 + k0, &Bs[buf][l1]);
    };

    v8f acc[2][4];
    #pragma unroll
    for (int i = 0; i < 2; i++)
        #pragma unroll
        for (int j = 0; j < 4; j++)
            #pragma unroll
            for (int e = 0; e < 8; e++) acc[i][j][e] = 0.f;

    const int klo  = (lane >> 4) << 3;   // 0 or 8 (ISA A/B frag layout)
    const int mrow = lane & 15;

    // prologue: stage first tile
    stage(0, 0);
    wait_async_all();
    __syncthreads();

    int buf = 0;
    for (int k0 = 0; k0 < K; k0 += 32) {
        if (k0 + 32 < K) stage(buf ^ 1, k0 + 32);   // DMA next tile during compute

        FragBF af[2], bf[4];
        #pragma unroll
        for (int i = 0; i < 2; i++) {
            const int r = wm * 32 + i * 16 + mrow;
            af[i].u[0] = *(const uint4*)&As[buf][r * 40 + klo];
            af[i].u[1] = *(const uint4*)&As[buf][r * 40 + klo + 16];
        }
        #pragma unroll
        for (int j = 0; j < 4; j++) {
            const int r = wn * 64 + j * 16 + mrow;
            bf[j].u[0] = *(const uint4*)&Bs[buf][r * 40 + klo];
            bf[j].u[1] = *(const uint4*)&Bs[buf][r * 40 + klo + 16];
        }
        #pragma unroll
        for (int i = 0; i < 2; i++)
            #pragma unroll
            for (int j = 0; j < 4; j++)
                acc[i][j] = __builtin_amdgcn_wmma_f32_16x16x32_bf16(
                    false, af[i].v, false, bf[j].v, (short)0, acc[i][j], false, false);

        wait_async_all();   // this wave's next-tile DMA complete
        __syncthreads();    // all waves: reads of buf done, writes of buf^1 visible
        buf ^= 1;
    }

    gemm_epilogue<MODE>(acc, Cp, ldc, M, N, bm, bn, wm, wn, lane, aux);
}

// ---------------------------------------------------------------- bf16 WMMA GEMM (guarded, for N<128)
template <int MODE>
__global__ void k_gemm_bf16(const ushort_t* __restrict__ A, int lda,
                            const ushort_t* __restrict__ B, int ldb,
                            void* __restrict__ Cp, int ldc,
                            int M, int N, int K,
                            const float* __restrict__ aux) {
    __shared__ ushort_t As[128 * 40];
    __shared__ ushort_t Bs[128 * 40];

    const int tid  = threadIdx.x;
    const int lane = tid & 31;
    const int wid  = tid >> 5;
    const int wm   = wid & 3;
    const int wn   = wid >> 2;
    const int bm   = blockIdx.y * 128;
    const int bn   = blockIdx.x * 128;

    v8f acc[2][4];
    #pragma unroll
    for (int i = 0; i < 2; i++)
        #pragma unroll
        for (int j = 0; j < 4; j++)
            #pragma unroll
            for (int e = 0; e < 8; e++) acc[i][j][e] = 0.f;

    const int klo  = (lane >> 4) << 3;
    const int mrow = lane & 15;

    for (int k0 = 0; k0 < K; k0 += 32) {
        #pragma unroll
        for (int c = tid; c < 512; c += 256) {
            const int r  = c >> 2;
            const int c8 = (c & 3) << 3;
            uint4 av = make_uint4(0, 0, 0, 0);
            uint4 bv = make_uint4(0, 0, 0, 0);
            const int gm = bm + r, gn = bn + r;
            if (gm < M) av = *(const uint4*)(A + (size_t)gm * lda + k0 + c8);
            if (gn < N) bv = *(const uint4*)(B + (size_t)gn * ldb + k0 + c8);
            *(uint4*)&As[r * 40 + c8] = av;
            *(uint4*)&Bs[r * 40 + c8] = bv;
        }
        __syncthreads();

        FragBF af[2], bf[4];
        #pragma unroll
        for (int i = 0; i < 2; i++) {
            const int r = wm * 32 + i * 16 + mrow;
            af[i].u[0] = *(const uint4*)&As[r * 40 + klo];
            af[i].u[1] = *(const uint4*)&As[r * 40 + klo + 16];
        }
        #pragma unroll
        for (int j = 0; j < 4; j++) {
            const int r = wn * 64 + j * 16 + mrow;
            bf[j].u[0] = *(const uint4*)&Bs[r * 40 + klo];
            bf[j].u[1] = *(const uint4*)&Bs[r * 40 + klo + 16];
        }
        #pragma unroll
        for (int i = 0; i < 2; i++)
            #pragma unroll
            for (int j = 0; j < 4; j++)
                acc[i][j] = __builtin_amdgcn_wmma_f32_16x16x32_bf16(
                    false, af[i].v, false, bf[j].v, (short)0, acc[i][j], false, false);
        __syncthreads();
    }

    gemm_epilogue<MODE>(acc, Cp, ldc, M, N, bm, bn, wm, wn, lane, aux);
}

// ---------------------------------------------------------------- causal depthwise conv + SiLU
__global__ void k_conv_silu(const ushort_t* __restrict__ xproj_bf,
                            const float* __restrict__ conv_w,
                            const float* __restrict__ conv_b,
                            float* __restrict__ u_f, ushort_t* __restrict__ u_bf) {
    const int n = L_SEQ * DI;
    for (int i = blockIdx.x * blockDim.x + threadIdx.x; i < n; i += gridDim.x * blockDim.x) {
        const int d = i & (DI - 1);
        const int l = i >> 11;           // DI = 2048
        float acc = conv_b[d];
        #pragma unroll
        for (int k = 0; k < KCONV; k++) {
            const int t = l + k - (KCONV - 1);
            if (t >= 0)
                acc += bf2f(xproj_bf[(size_t)t * (2 * DI) + d]) * conv_w[d * KCONV + k];
        }
        const float s = acc / (1.f + __expf(-acc));   // SiLU
        u_f[i]  = s;
        u_bf[i] = f2bf(s);
    }
}

// ---------------------------------------------------------------- selective scan
// lane <-> (channel d, state n); y reduced over 16 state lanes via __shfl_xor.
// delta buffer overwritten in place with y (read-before-write per t, same wave).
__global__ void k_scan(const float* __restrict__ xdbl,       // [L, 96]: [dr | B | C]
                       float* __restrict__ delta_y,          // in: delta [L,DI]; out: y
                       const float* __restrict__ u_f,        // [L, DI]
                       const float* __restrict__ A_log,      // [DI, DS]
                       const float* __restrict__ Dvec) {     // [DI]
    const int g = blockIdx.x * blockDim.x + threadIdx.x;     // 0 .. DI*DS-1
    const int d = g >> 4;
    const int nst = g & 15;
    const float Aval = -__expf(A_log[(size_t)d * DS + nst]);
    const float Dv = Dvec[d];
    float h = 0.f;
    for (int t = 0; t < L_SEQ; t++) {
        const size_t base = (size_t)t * DI + d;
        const float delta = delta_y[base];
        const float uv    = u_f[base];
        const float Bv = xdbl[(size_t)t * XDBL_N + DTR + nst];
        const float Cv = xdbl[(size_t)t * XDBL_N + DTR + DS + nst];
        h = __expf(delta * Aval) * h + (delta * uv) * Bv;
        float p = h * Cv;
        p += __shfl_xor(p, 1, 32);
        p += __shfl_xor(p, 2, 32);
        p += __shfl_xor(p, 4, 32);
        p += __shfl_xor(p, 8, 32);
        if (nst == 0) delta_y[base] = p + uv * Dv;
    }
}

// ---------------------------------------------------------------- gating: g = y * silu(gate) -> bf16
__global__ void k_gate(const float* __restrict__ y, const ushort_t* __restrict__ xproj_bf,
                       ushort_t* __restrict__ g_bf) {
    const int n = L_SEQ * DI;
    for (int i = blockIdx.x * blockDim.x + threadIdx.x; i < n; i += gridDim.x * blockDim.x) {
        const int d = i & (DI - 1);
        const int l = i >> 11;
        const float gt = bf2f(xproj_bf[(size_t)l * (2 * DI) + DI + d]);
        const float sg = gt / (1.f + __expf(-gt));
        g_bf[i] = f2bf(y[i] * sg);
    }
}

// ================================================================ host launcher
extern "C" void kernel_launch(void* const* d_in, const int* in_sizes, int n_in,
                              void* d_out, int out_size, void* d_ws, size_t ws_size,
                              hipStream_t stream) {
    const float* x       = (const float*)d_in[0];
    const float* norm_w  = (const float*)d_in[1];
    const float* norm_b  = (const float*)d_in[2];
    const float* W_in    = (const float*)d_in[3];
    const float* conv_w  = (const float*)d_in[4];
    const float* conv_b  = (const float*)d_in[5];
    const float* W_xproj = (const float*)d_in[6];
    const float* W_dt    = (const float*)d_in[7];
    const float* b_dt    = (const float*)d_in[8];
    const float* A_log   = (const float*)d_in[9];
    const float* Dvec    = (const float*)d_in[10];
    const float* W_out   = (const float*)d_in[11];
    float* out = (float*)d_out;

    char* ws = (char*)d_ws;
    size_t off = 0;
    auto alloc = [&](size_t bytes) -> char* {
        char* p = ws + off;
        off = (off + bytes + 255) & ~(size_t)255;
        return p;
    };
    ushort_t* win_bf   = (ushort_t*)alloc((size_t)2 * DI * DM * 2);       // [4096,1024]
    ushort_t* wxp_bf   = (ushort_t*)alloc((size_t)XDBL_N * DI * 2);       // [96,2048]
    ushort_t* wdt_bf   = (ushort_t*)alloc((size_t)DI * DTR * 2);          // [2048,64]
    ushort_t* wout_bf  = (ushort_t*)alloc((size_t)DM * DI * 2);           // [1024,2048]
    ushort_t* xnorm_bf = (ushort_t*)alloc((size_t)L_SEQ * DM * 2);        // [4096,1024]
    ushort_t* xproj_bf = (ushort_t*)alloc((size_t)L_SEQ * 2 * DI * 2);    // [4096,4096]
    float*    u_f      = (float*)alloc((size_t)L_SEQ * DI * 4);           // [4096,2048]
    ushort_t* u_bf     = (ushort_t*)alloc((size_t)L_SEQ * DI * 2);        // reused as g_bf
    float*    xdbl     = (float*)alloc((size_t)L_SEQ * XDBL_N * 4);       // [4096,96]
    ushort_t* dtr_bf   = (ushort_t*)alloc((size_t)L_SEQ * DTR * 2);       // [4096,64]
    float*    delta_y  = (float*)alloc((size_t)L_SEQ * DI * 4);           // delta -> y in place
    (void)ws_size; (void)in_sizes; (void)n_in; (void)out_size;

    // 1. weight conversion f32 -> bf16
    k_f2bf<<<2048, 256, 0, stream>>>(W_in,    win_bf,  2 * DI * DM);
    k_f2bf<<<512,  256, 0, stream>>>(W_xproj, wxp_bf,  XDBL_N * DI);
    k_f2bf<<<512,  256, 0, stream>>>(W_dt,    wdt_bf,  DI * DTR);
    k_f2bf<<<2048, 256, 0, stream>>>(W_out,   wout_bf, DM * DI);

    // 2. LayerNorm -> bf16
    k_layernorm<<<L_SEQ, 256, 0, stream>>>(x, norm_w, norm_b, xnorm_bf);

    // 3. in-proj: [4096,4096] = xnorm @ W_in^T   (store bf16)
    k_gemm_bf16_dbuf<1><<<dim3(2 * DI / 128, L_SEQ / 128), 256, 0, stream>>>(
        xnorm_bf, DM, win_bf, DM, xproj_bf, 2 * DI, L_SEQ, 2 * DI, DM, nullptr);

    // 4. causal depthwise conv + SiLU -> u (f32 + bf16)
    k_conv_silu<<<L_SEQ * DI / 256, 256, 0, stream>>>(xproj_bf, conv_w, conv_b, u_f, u_bf);

    // 5. x-proj: [4096,96] = u @ W_xproj^T   (store f32; N<128 -> guarded kernel)
    k_gemm_bf16<0><<<dim3(1, L_SEQ / 128), 256, 0, stream>>>(
        u_bf, DI, wxp_bf, DI, xdbl, XDBL_N, L_SEQ, XDBL_N, DI, nullptr);

    // 6. extract delta_r as bf16
    k_f2bf_strided<<<1024, 256, 0, stream>>>(xdbl, dtr_bf, L_SEQ, DTR, XDBL_N, DTR);

    // 7. delta-proj with fused bias + softplus: [4096,2048]
    k_gemm_bf16_dbuf<2><<<dim3(DI / 128, L_SEQ / 128), 256, 0, stream>>>(
        dtr_bf, DTR, wdt_bf, DTR, delta_y, DI, L_SEQ, DI, DTR, b_dt);

    // 8. selective scan (writes y in place over delta)
    k_scan<<<DI * DS / 256, 256, 0, stream>>>(xdbl, delta_y, u_f, A_log, Dvec);

    // 9. gating -> bf16 (reuses u_bf buffer)
    k_gate<<<L_SEQ * DI / 256, 256, 0, stream>>>(delta_y, xproj_bf, u_bf);

    // 10. out-proj with fused residual add: out = x + g @ W_out^T
    k_gemm_bf16_dbuf<3><<<dim3(DM / 128, L_SEQ / 128), 256, 0, stream>>>(
        u_bf, DI, wout_bf, DI, out, DM, L_SEQ, DM, DI, x);
}
